// NETWORK_17592186044755
// MI455X (gfx1250) — matrix operand order repaired and checked by reading, not compiled, not run on hardware
//
#include <hip/hip_runtime.h>

// CDNA5 (gfx1250) wave32 WMMA types
typedef __attribute__((ext_vector_type(16))) _Float16 v16h;
typedef __attribute__((ext_vector_type(8)))  float    v8f;
typedef __attribute__((ext_vector_type(4)))  int      v4i;

#define HDIM 64
#define KDIM 256
#define RATE 0.999f
#define EMA_EPS 1e-6f

#define TOK_PER_BLK 32      // 2 waves x 16 token rows
#define THREADS 64
#define XSTR 68             // padded LDS stride (floats) for x / out tile
#define MNSTR 72            // padded LDS stride (halfs): 144B rows, 16B-aligned chunks for async B128
#define SLSTR 264           // padded LDS stride (halfs) for soft-label tile

// ---------------------------------------------------------------------------
// Typed address-space pointers for the gfx1250 async Global<->LDS builtins:
//   global_load_async_to_lds_b128(v4i AS1*, v4i AS3*, Ii offset, Ii cpol)
//   global_load_async_to_lds_b32 (int AS1*, int AS3*, Ii, Ii)
//   global_store_async_from_lds_b32(int AS1*, int AS3*, Ii, Ii)
// Generic LDS pointers carry the LDS offset in the low 32 bits (ISA 10.2),
// so integer truncation is the portable generic->AS3 conversion.
// ---------------------------------------------------------------------------
typedef __attribute__((address_space(1))) v4i* g_v4i_p;
typedef __attribute__((address_space(3))) v4i* l_v4i_p;
typedef __attribute__((address_space(1))) int* g_i32_p;
typedef __attribute__((address_space(3))) int* l_i32_p;

__device__ __forceinline__ g_v4i_p gbl_v4i(const void* p) {
    return (g_v4i_p)(unsigned long long)p;
}
__device__ __forceinline__ l_v4i_p lds_v4i(const void* p) {
    return (l_v4i_p)(unsigned int)(unsigned long long)p;
}
__device__ __forceinline__ g_i32_p gbl_i32(const void* p) {
    return (g_i32_p)(unsigned long long)p;
}
__device__ __forceinline__ l_i32_p lds_i32(const void* p) {
    return (l_i32_p)(unsigned int)(unsigned long long)p;
}

__device__ __forceinline__ void async_ld_b128(const void* g, void* l) {
#if __has_builtin(__builtin_amdgcn_global_load_async_to_lds_b128)
    __builtin_amdgcn_global_load_async_to_lds_b128(gbl_v4i(g), lds_v4i(l), 0, 0);
#else
    *(uint4*)l = *(const uint4*)g;
#endif
}
__device__ __forceinline__ void async_ld_b32(const void* g, void* l) {
#if __has_builtin(__builtin_amdgcn_global_load_async_to_lds_b32)
    __builtin_amdgcn_global_load_async_to_lds_b32(gbl_i32(g), lds_i32(l), 0, 0);
#else
    *(float*)l = *(const float*)g;
#endif
}
__device__ __forceinline__ void async_st_b32(void* g, const void* l) {
#if __has_builtin(__builtin_amdgcn_global_store_async_from_lds_b32)
    __builtin_amdgcn_global_store_async_from_lds_b32(gbl_i32(g), lds_i32(l), 0, 0);
#else
    *(float*)g = *(const float*)l;
#endif
}
__device__ __forceinline__ void wait_async() {
#if __has_builtin(__builtin_amdgcn_s_wait_asynccnt)
    __builtin_amdgcn_s_wait_asynccnt(0);
#else
    asm volatile("s_wait_asynccnt 0" ::: "memory");
#endif
}

// ---------------------------------------------------------------------------
// Branchless cross-lane reductions over the 16 lanes that hold one C-row,
// via ds_swizzle_b32 group-of-32 XOR patterns: offset = (xor_mask<<10)|0x1f.
// NOTE: bitwise |,& (not ||,&&) so clang lowers to v_cmp + cndmask, no
// saveexec branching.
// ---------------------------------------------------------------------------
template <int PAT>
__device__ __forceinline__ void argmax_merge(float (&bv)[8], int (&bi)[8]) {
#pragma unroll
    for (int j = 0; j < 8; ++j) {
        const float ov = __int_as_float(__builtin_amdgcn_ds_swizzle(__float_as_int(bv[j]), PAT));
        const int   oi = __builtin_amdgcn_ds_swizzle(bi[j], PAT);
        const int gt = (int)(ov > bv[j]);
        const int tb = (int)(ov == bv[j]) & (int)(oi < bi[j]);
        const bool better = (gt | tb) != 0;
        bv[j] = better ? ov : bv[j];
        bi[j] = better ? oi : bi[j];
    }
}
template <int PAT>
__device__ __forceinline__ void fmax_merge(float (&v)[8]) {
#pragma unroll
    for (int j = 0; j < 8; ++j)
        v[j] = fmaxf(v[j], __int_as_float(__builtin_amdgcn_ds_swizzle(__float_as_int(v[j]), PAT)));
}
template <int PAT>
__device__ __forceinline__ void fsum_merge(float (&v)[8]) {
#pragma unroll
    for (int j = 0; j < 8; ++j)
        v[j] += __int_as_float(__builtin_amdgcn_ds_swizzle(__float_as_int(v[j]), PAT));
}
#define SWZ_X1 0x041f
#define SWZ_X2 0x081f
#define SWZ_X4 0x101f
#define SWZ_X8 0x201f

// ---------------------------------------------------------------------------
// EMA codebook update + renormalize. One block of KDIM threads, thread = code.
// init==1: m = units (first pass). Writes m (f32), mn (f16, unit rows), |m|.
// ---------------------------------------------------------------------------
__global__ __launch_bounds__(KDIM)
void vq_update_kernel(const float* __restrict__ units,
                      float* __restrict__ m,
                      _Float16* __restrict__ mn,
                      float* __restrict__ norms,
                      const float* __restrict__ embed_sum,
                      const float* __restrict__ counts,
                      int init)
{
    const int r = threadIdx.x;
    float row[HDIM];
    if (init) {
        for (int i = 0; i < HDIM; ++i) row[i] = units[r * HDIM + i];
    } else {
        const float cinv = (1.0f - RATE) / (counts[r] + EMA_EPS);
        for (int i = 0; i < HDIM; ++i)
            row[i] = m[r * HDIM + i] * RATE + embed_sum[r * HDIM + i] * cinv;
    }
    float s = 0.0f;
    for (int i = 0; i < HDIM; ++i) { m[r * HDIM + i] = row[i]; s += row[i] * row[i]; }
    const float nrm = sqrtf(s);
    norms[r] = nrm;
    const float rinv = 1.0f / fmaxf(nrm, 1e-12f);
    for (int i = 0; i < HDIM; ++i) mn[r * HDIM + i] = (_Float16)(row[i] * rinv);
}

// ---------------------------------------------------------------------------
// Shared staging: codebook tile via async B128, x tile via async B32.
// ---------------------------------------------------------------------------
__device__ __forceinline__ void stage_mn(const _Float16* __restrict__ mn,
                                         _Float16* mn_lds, int tid)
{
    // 256 rows x 8 chunks of 8 halfs (16B); 2048 chunks over 64 threads.
#pragma unroll 4
    for (int c = tid; c < KDIM * (HDIM / 8); c += THREADS) {
        const int r  = c >> 3;
        const int ck = (c & 7) * 8;
        async_ld_b128(mn + r * HDIM + ck, &mn_lds[r * MNSTR + ck]);
    }
}

__device__ __forceinline__ void stage_x(const float* __restrict__ xb,
                                        float* x_lds, int tid, int hw)
{
    const int tl  = tid & 31;   // local token (spatial offset)
    const int chh = tid >> 5;   // channel phase
#pragma unroll
    for (int step = 0; step < HDIM / 2; ++step) {
        const int ch = step * 2 + chh;
        async_ld_b32(xb + (long)ch * hw + tl, &x_lds[tl * XSTR + ch]);
    }
}

// ---------------------------------------------------------------------------
// Assignment pass: score = xn @ mn^T via v_wmma_f32_16x16x32_f16, per-row
// argmax over K=256, scatter-sum raw x rows + counts with global f32 atomics.
// Block: 64 threads (2 waves), 32 tokens.
// ---------------------------------------------------------------------------
__global__ __launch_bounds__(THREADS)
void vq_assign_kernel(const float* __restrict__ x,
                      const _Float16* __restrict__ mn,
                      float* __restrict__ embed_sum,
                      float* __restrict__ counts,
                      int hw)
{
    __shared__ _Float16 mn_lds[KDIM * MNSTR];
    __shared__ float    x_lds[TOK_PER_BLK * XSTR];
    __shared__ float    rn_lds[TOK_PER_BLK];
    __shared__ int      idx_lds[TOK_PER_BLK];

    const int tid  = threadIdx.x;
    const int lane = tid & 31;
    const int wave = tid >> 5;

    const long t0    = (long)blockIdx.x * TOK_PER_BLK;
    const int  batch = (int)(t0 / hw);
    const int  s0    = (int)(t0 % hw);
    const float* xb  = x + ((long)batch * HDIM) * (long)hw + s0;

    stage_mn(mn, mn_lds, tid);
    stage_x(xb, x_lds, tid, hw);
    wait_async();
    __syncthreads();

    // ---- per-token inverse l2 norm ----
    if (tid < TOK_PER_BLK) {
        float s = 0.0f;
        for (int i = 0; i < HDIM; ++i) { const float v = x_lds[tid * XSTR + i]; s += v * v; }
        rn_lds[tid] = 1.0f / fmaxf(sqrtf(s), 1e-12f);
    }
    __syncthreads();

    // ---- build A fragments (16-bit A 16x32 layout), K = channels 0..63 ----
    const int halfsel = (lane < 16) ? 0 : 1;
    const int row     = (wave << 4) + (lane & 15);
    const float rn    = rn_lds[row];
    const int klo     = halfsel ? 8 : 0;
    v16h a0, a1;
#pragma unroll
    for (int i = 0; i < 8; ++i) {
        a0[i]     = (_Float16)(x_lds[row * XSTR + klo + i]            * rn);
        a0[8 + i] = (_Float16)(x_lds[row * XSTR + klo + 16 + i]       * rn);
        a1[i]     = (_Float16)(x_lds[row * XSTR + 32 + klo + i]       * rn);
        a1[8 + i] = (_Float16)(x_lds[row * XSTR + 32 + klo + 16 + i]  * rn);
    }

    // ---- sweep 16 N-tiles of codes, track per-row argmax (branchless) ----
    float bestv[8];
    int   besti[8];
#pragma unroll
    for (int j = 0; j < 8; ++j) { bestv[j] = -3.4e38f; besti[j] = 0; }
    const int col = lane & 15;
    const int kb  = halfsel ? 8 : 0;    // 16-bit B 32x16: lanes 0-15 K0..7,16..23; 16-31 K8..15,24..31

    for (int nt = 0; nt < KDIM / 16; ++nt) {
        v16h b0, b1;
        const _Float16* mrow = &mn_lds[(nt * 16 + col) * MNSTR];
#pragma unroll
        for (int i = 0; i < 16; ++i) { b0[i] = mrow[kb + (i & 7) + ((i >> 3) << 4)]; b1[i] = mrow[32 + kb + (i & 7) + ((i >> 3) << 4)]; }
        v8f c = {};
        c = __builtin_amdgcn_wmma_f32_16x16x32_f16(false, a0, false, b0, (short)0, c, false, false);
        c = __builtin_amdgcn_wmma_f32_16x16x32_f16(false, a1, false, b1, (short)0, c, false, false);
        const int code = nt * 16 + col;
#pragma unroll
        for (int j = 0; j < 8; ++j) {
            const float v = c[j];
            const bool better = v > bestv[j];
            besti[j] = better ? code : besti[j];
            bestv[j] = better ? v : bestv[j];
        }
    }

    // ---- butterfly argmax across the 16 lanes holding each row ----
    argmax_merge<SWZ_X1>(bestv, besti);
    argmax_merge<SWZ_X2>(bestv, besti);
    argmax_merge<SWZ_X4>(bestv, besti);
    argmax_merge<SWZ_X8>(bestv, besti);

    if ((lane & 15) == 0) {
        const int rbase = (wave << 4) + (halfsel ? 8 : 0);
        for (int j = 0; j < 8; ++j) idx_lds[rbase + j] = besti[j];
    }
    __syncthreads();

    // ---- scatter-sum: counts + per-channel coalesced f32 atomics ----
    if (tid < TOK_PER_BLK) atomicAdd(&counts[idx_lds[tid]], 1.0f);
#pragma unroll 4
    for (int t = 0; t < TOK_PER_BLK; ++t)
        atomicAdd(&embed_sum[idx_lds[t] * HDIM + tid], x_lds[t * XSTR + tid]);
}

// ---------------------------------------------------------------------------
// Final pass: scores (WMMA) -> softmax over K=256 -> out = (p*|m|) @ mn (WMMA)
// The |m| fold lets both GEMMs share the same f16 mn tile in LDS.
// ---------------------------------------------------------------------------
__global__ __launch_bounds__(THREADS)
void vq_output_kernel(const float* __restrict__ x,
                      const _Float16* __restrict__ mn,
                      const float* __restrict__ norms,
                      float* __restrict__ out,
                      int hw)
{
    __shared__ _Float16 mn_lds[KDIM * MNSTR];
    __shared__ float    xo_lds[TOK_PER_BLK * XSTR];   // x tile, later reused for out tile
    __shared__ float    rn_lds[TOK_PER_BLK];
    __shared__ float    norm_lds[KDIM];
    __shared__ _Float16 sl_lds[TOK_PER_BLK * SLSTR];  // soft labels * |m|, f16

    const int tid  = threadIdx.x;
    const int lane = tid & 31;
    const int wave = tid >> 5;

    const long t0    = (long)blockIdx.x * TOK_PER_BLK;
    const int  batch = (int)(t0 / hw);
    const int  s0    = (int)(t0 % hw);
    const float* xb  = x + ((long)batch * HDIM) * (long)hw + s0;

    stage_mn(mn, mn_lds, tid);
    stage_x(xb, xo_lds, tid, hw);
    for (int r = tid; r < KDIM; r += THREADS) norm_lds[r] = norms[r];
    wait_async();
    __syncthreads();

    if (tid < TOK_PER_BLK) {
        float s = 0.0f;
        for (int i = 0; i < HDIM; ++i) { const float v = xo_lds[tid * XSTR + i]; s += v * v; }
        rn_lds[tid] = 1.0f / fmaxf(sqrtf(s), 1e-12f);
    }
    __syncthreads();

    const int halfsel = (lane < 16) ? 0 : 1;
    const int row     = (wave << 4) + (lane & 15);
    const float rn    = rn_lds[row];
    const int klo     = halfsel ? 8 : 0;
    const int kb      = halfsel ? 8 : 0;
    const int col     = lane & 15;

    v16h a0, a1;
#pragma unroll
    for (int i = 0; i < 8; ++i) {
        a0[i]     = (_Float16)(xo_lds[row * XSTR + klo + i]           * rn);
        a0[8 + i] = (_Float16)(xo_lds[row * XSTR + klo + 16 + i]      * rn);
        a1[i]     = (_Float16)(xo_lds[row * XSTR + 32 + klo + i]      * rn);
        a1[8 + i] = (_Float16)(xo_lds[row * XSTR + 32 + klo + 16 + i] * rn);
    }

    // ---- all 256 scores per row live in 16 v8f accumulators ----
    v8f sc[KDIM / 16];
    for (int nt = 0; nt < KDIM / 16; ++nt) {
        v16h b0, b1;
        const _Float16* mrow = &mn_lds[(nt * 16 + col) * MNSTR];
#pragma unroll
        for (int i = 0; i < 16; ++i) { b0[i] = mrow[kb + (i & 7) + ((i >> 3) << 4)]; b1[i] = mrow[32 + kb + (i & 7) + ((i >> 3) << 4)]; }
        v8f c = {};
        c = __builtin_amdgcn_wmma_f32_16x16x32_f16(false, a0, false, b0, (short)0, c, false, false);
        c = __builtin_amdgcn_wmma_f32_16x16x32_f16(false, a1, false, b1, (short)0, c, false, false);
        sc[nt] = c;
    }

    // ---- softmax over K: ds_swizzle butterfly max / sum across 16 lanes ----
    float rmax[8];
#pragma unroll
    for (int j = 0; j < 8; ++j) rmax[j] = -3.4e38f;
    for (int nt = 0; nt < KDIM / 16; ++nt)
#pragma unroll
        for (int j = 0; j < 8; ++j) rmax[j] = fmaxf(rmax[j], sc[nt][j]);
    fmax_merge<SWZ_X1>(rmax);
    fmax_merge<SWZ_X2>(rmax);
    fmax_merge<SWZ_X4>(rmax);
    fmax_merge<SWZ_X8>(rmax);

    float rsum[8];
#pragma unroll
    for (int j = 0; j < 8; ++j) rsum[j] = 0.0f;
    for (int nt = 0; nt < KDIM / 16; ++nt)
#pragma unroll
        for (int j = 0; j < 8; ++j) {
            const float e = __expf(sc[nt][j] - rmax[j]);
            sc[nt][j] = e;
            rsum[j] += e;
        }
    fsum_merge<SWZ_X1>(rsum);
    fsum_merge<SWZ_X2>(rsum);
    fsum_merge<SWZ_X4>(rsum);
    fsum_merge<SWZ_X8>(rsum);

    // ---- write p * |m_code| into A-layout staging tile ----
    {
        const int rbase = (wave << 4) + (halfsel ? 8 : 0);
        for (int nt = 0; nt < KDIM / 16; ++nt) {
            const int code = nt * 16 + col;
            const float nk = norm_lds[code];
#pragma unroll
            for (int j = 0; j < 8; ++j) {
                const float p = sc[nt][j] / rsum[j];
                sl_lds[(rbase + j) * SLSTR + code] = (_Float16)(p * nk);
            }
        }
    }
    __syncthreads();

    // ---- second GEMM: out[16 x 64] = SL[16 x 256] @ mn[256 x 64] ----
    for (int nt2 = 0; nt2 < HDIM / 16; ++nt2) {
        v8f d = {};
        for (int ks = 0; ks < KDIM / 32; ++ks) {
            v16h af, bf;
            const _Float16* sr = &sl_lds[row * SLSTR + ks * 32];
#pragma unroll
            for (int i = 0; i < 8; ++i) { af[i] = sr[klo + i]; af[8 + i] = sr[klo + 16 + i]; }
            const int ch = nt2 * 16 + col;
#pragma unroll
            for (int i = 0; i < 16; ++i) bf[i] = mn_lds[(ks * 32 + kb + (i & 7) + ((i >> 3) << 4)) * MNSTR + ch];
            d = __builtin_amdgcn_wmma_f32_16x16x32_f16(false, af, false, bf, (short)0, d, false, false);
        }
        const int rbase = (wave << 4) + (halfsel ? 8 : 0);
#pragma unroll
        for (int j = 0; j < 8; ++j)
            xo_lds[(rbase + j) * XSTR + nt2 * 16 + col] = d[j];   // reuse x tile as out tile
    }
    __syncthreads();

    // ---- async store out tile from LDS, channel-major coalesced ----
    {
        const int tl  = tid & 31;
        const int chh = tid >> 5;
        float* ob = out + ((long)batch * HDIM) * (long)hw + s0;
#pragma unroll
        for (int step = 0; step < HDIM / 2; ++step) {
            const int ch = step * 2 + chh;
            async_st_b32(ob + (long)ch * hw + tl, &xo_lds[tl * XSTR + ch]);
        }
        wait_async();
    }
}

// ---------------------------------------------------------------------------
extern "C" void kernel_launch(void* const* d_in, const int* in_sizes, int n_in,
                              void* d_out, int out_size, void* d_ws, size_t ws_size,
                              hipStream_t stream)
{
    const float* x     = (const float*)d_in[0];  // [4, 64, 256, 256] f32
    const float* units = (const float*)d_in[1];  // [256, 64] f32
    float* out = (float*)d_out;

    const int n  = in_sizes[0] / HDIM;        // 262144 tokens
    const int hw = in_sizes[0] / (HDIM * 4);  // 65536 spatial per batch (b=4)

    // workspace layout
    char* ws = (char*)d_ws;
    float*    m_ws     = (float*)(ws);                                  // 64 KB
    _Float16* mn_ws    = (_Float16*)(ws + 65536);                       // 32 KB
    float*    norms_ws = (float*)(ws + 65536 + 32768);                  // 1 KB (pad to 4 KB)
    float*    esum_ws  = (float*)(ws + 65536 + 32768 + 4096);           // 64 KB
    float*    cnt_ws   = (float*)(ws + 65536 + 32768 + 4096 + 65536);   // 1 KB

    const int nblk = n / TOK_PER_BLK;

    // m0 = units; mn0 = l2norm(m0)
    vq_update_kernel<<<1, KDIM, 0, stream>>>(units, m_ws, mn_ws, norms_ws, esum_ws, cnt_ws, 1);

    for (int it = 0; it < 3; ++it) {
        (void)hipMemsetAsync(esum_ws, 0, KDIM * HDIM * sizeof(float), stream);
        (void)hipMemsetAsync(cnt_ws, 0, KDIM * sizeof(float), stream);
        vq_assign_kernel<<<nblk, THREADS, 0, stream>>>(x, mn_ws, esum_ws, cnt_ws, hw);
        vq_update_kernel<<<1, KDIM, 0, stream>>>(units, m_ws, mn_ws, norms_ws, esum_ws, cnt_ws, 0);
    }

    vq_output_kernel<<<nblk, THREADS, 0, stream>>>(x, mn_ws, norms_ws, out, hw);
}
